// SDFNet_40037685133508
// MI455X (gfx1250) — matrix operand order repaired and checked by reading, not compiled, not run on hardware
//
#include <hip/hip_runtime.h>

typedef __attribute__((ext_vector_type(16))) _Float16 v16h;
typedef __attribute__((ext_vector_type(8)))  _Float16 v8h;
typedef __attribute__((ext_vector_type(2)))  _Float16 h2;
typedef __attribute__((ext_vector_type(8)))  float    v8f;

#define T_SIZE (1u << 19)
#define TMASK  (T_SIZE - 1u)

#define ENC_STRIDE 40   // 32 feature halves + 8 pad halves (80 B rows, 16B-aligned, bank-spread)
#define H_STRIDE   72   // 64 hidden halves + 8 pad halves (144 B rows)

union VA { v16h v; v8h h[2]; };

// Branch-free softplus_beta100 on raw hardware transcendentals (bare v_exp_f32/v_log_f32,
// both base-2). Guards are provably unneeded: exp2 arg <= 20*log2e < 29 (no range fixup),
// log2 arg >= 1 (no denorm fixup, no cancellation).
//   z = 100x; z > 20 -> x, else log(1+exp(z))/100
//     exp(z)      = exp2(z * log2e)           -> fold 100*log2e into one constant
//     log(1+e)/100 = log2(1+e) * (ln2/100)
__device__ __forceinline__ float softplus100(float x)
{
    const float LOG2E_100 = 144.269504088896340736f;   // 100 * log2(e)
    const float LN2_001   = 0.00693147180559945309f;   // ln(2) / 100
    const float z2 = fminf(LOG2E_100 * x, 28.8539008177792681472f); // min(z,20)*log2e
    const float e  = __builtin_amdgcn_exp2f(z2);
    const float sp = __builtin_amdgcn_logf(1.0f + e) * LN2_001;
    return (x > 0.20f) ? x : sp;     // z > 20  <=>  x > 0.2 ; lowers to v_cndmask
}

__global__ __launch_bounds__(128)
void sdfnet_fused(const float* __restrict__ x,
                  const float* __restrict__ table,
                  const float* __restrict__ W0,
                  const float* __restrict__ b0,
                  const float* __restrict__ W1,
                  const float* __restrict__ b1,
                  float* __restrict__ out,
                  int npts)
{
    __shared__ _Float16 sW0[64 * 32];          // layer0 weights, row-major [n][k]
    __shared__ _Float16 sW1[16 * 64];          // layer1 weights, row-major [n][k]
    __shared__ float    sB0[64];
    __shared__ float    sB1[16];
    __shared__ _Float16 sEnc[4][32 * ENC_STRIDE]; // per-wave 32 points x 32 feats
    __shared__ _Float16 sH[4][16 * H_STRIDE];     // per-wave 16 points x 64 hidden

    const int tid  = threadIdx.x;
    const int wave = tid >> 5;
    const int lane = tid & 31;
    const int lm   = lane & 15;
    const int hi   = lane & 16;        // 0 for lanes 0-15, 16 for lanes 16-31

    // ---- stage weights into LDS as f16 ----
    for (int i = tid; i < 64 * 32; i += 128) sW0[i] = (_Float16)W0[i];
    for (int i = tid; i < 16 * 64; i += 128) sW1[i] = (_Float16)W1[i];
    if (tid < 64) sB0[tid] = b0[tid];
    if (tid < 16) sB1[tid] = b1[tid];

    // ---- hash-grid encode: one point per lane (32 points / wave) ----
    {
        const int point = blockIdx.x * 128 + tid;
        const float xr = x[3 * point + 0];
        const float yr = x[3 * point + 1];
        const float zr = x[3 * point + 2];
        const float x0 = fminf(fmaxf((xr + 1.0f) * 0.5f, 0.0f), 1.0f);
        const float y0 = fminf(fmaxf((yr + 1.0f) * 0.5f, 0.0f), 1.0f);
        const float z0 = fminf(fmaxf((zr + 1.0f) * 0.5f, 0.0f), 1.0f);

        #pragma unroll 1
        for (int l = 0; l < 16; ++l) {
            // scale = 16 * 2^(7l/15) - 1
            const float scale = exp2f((float)l * (7.0f / 15.0f) + 4.0f) - 1.0f;
            const unsigned res = (unsigned)ceilf(scale) + 1u;
            const bool dense = ((unsigned long long)res * res * res) <= (unsigned long long)T_SIZE;

            const float px = x0 * scale + 0.5f;
            const float py = y0 * scale + 0.5f;
            const float pz = z0 * scale + 0.5f;
            const float gx = floorf(px), gy = floorf(py), gz = floorf(pz);
            const float fx = px - gx, fy = py - gy, fz = pz - gz;
            const unsigned ix = (unsigned)gx, iy = (unsigned)gy, iz = (unsigned)gz;

            const float* lt = table + (size_t)l * (size_t)T_SIZE * 2u;
            float s0 = 0.0f, s1 = 0.0f;
            #pragma unroll
            for (int c = 0; c < 8; ++c) {
                const unsigned cx = ix + (c & 1);
                const unsigned cy = iy + ((c >> 1) & 1);
                const unsigned cz = iz + (c >> 2);
                unsigned idx = dense
                    ? (cx + cy * res + cz * (res * res))
                    : (cx ^ (cy * 2654435761u) ^ (cz * 805459861u));
                idx &= TMASK;
                const float2 e = *(const float2*)(lt + 2u * idx);   // L2-resident gather
                const float w = ((c & 1)        ? fx : 1.0f - fx)
                              * (((c >> 1) & 1) ? fy : 1.0f - fy)
                              * ((c >> 2)       ? fz : 1.0f - fz);
                s0 = fmaf(w, e.x, s0);
                s1 = fmaf(w, e.y, s1);
            }
            h2 fv; fv.x = (_Float16)s0; fv.y = (_Float16)s1;
            *(h2*)&sEnc[wave][lane * ENC_STRIDE + 2 * l] = fv;      // ds_store_b32
        }
    }
    __syncthreads();

    // ---- MLP: two 16-point WMMA tiles per wave ----
    #pragma unroll 1
    for (int t = 0; t < 2; ++t) {
        // A operand (enc tile, 16x32 f16):
        // lanes 0-15 hold row M=lane, K={0..7,16..23}; lanes 16-31 hold K={8..15,24..31}
        VA a;
        {
            const int row = t * 16 + lm;
            const int k0  = hi ? 8 : 0;
            const _Float16* base = &sEnc[wave][row * ENC_STRIDE];
            a.h[0] = *(const v8h*)(base + k0);
            a.h[1] = *(const v8h*)(base + k0 + 16);
        }

        // Layer 0: hidden = softplus_beta100(enc @ W0^T + b0), 4 N-tiles of 16
        #pragma unroll
        for (int nt = 0; nt < 4; ++nt) {
            // B operand (32x16): lane holds column n=lane%16, K={0..7,16..23} / {8..15,24..31}
            VA bmat;
            const int n  = nt * 16 + lm;
            const int kb = hi ? 8 : 0;
            const _Float16* wb = &sW0[n * 32 + kb];
            bmat.h[0] = *(const v8h*)(wb);
            bmat.h[1] = *(const v8h*)(wb + 16);

            const float bias = sB0[n];
            v8f c = {bias, bias, bias, bias, bias, bias, bias, bias};
            c = __builtin_amdgcn_wmma_f32_16x16x32_f16(
                    false, a.v, false, bmat.v, (short)0, c, false, false);

            // D layout: VGPR j, lane -> (M = j + (hi?8:0), N = n); softplus + f16 -> LDS
            #pragma unroll
            for (int j = 0; j < 8; ++j) {
                const float sp = softplus100(c[j]);
                const int m = j + (hi ? 8 : 0);
                sH[wave][m * H_STRIDE + n] = (_Float16)sp;
            }
        }
        __syncthreads();

        // Layer 1: out = h @ W1^T + b1   (K=64 -> two K-steps of 32, single N-tile)
        const float bias1 = sB1[lm];
        v8f c1 = {bias1, bias1, bias1, bias1, bias1, bias1, bias1, bias1};
        #pragma unroll
        for (int kb = 0; kb < 2; ++kb) {
            VA a1, b1m;
            const int koff = kb * 32 + (hi ? 8 : 0);
            const _Float16* ab = &sH[wave][lm * H_STRIDE];
            a1.h[0] = *(const v8h*)(ab + koff);
            a1.h[1] = *(const v8h*)(ab + koff + 16);

            const int kw = kb * 32 + (hi ? 8 : 0);
            const _Float16* wb = &sW1[lm * 64 + kw];
            b1m.h[0] = *(const v8h*)(wb);
            b1m.h[1] = *(const v8h*)(wb + 16);

            c1 = __builtin_amdgcn_wmma_f32_16x16x32_f16(
                    false, a1.v, false, b1m.v, (short)0, c1, false, false);
        }

        // Scatter D: n==0 -> sdf[pt]; n>0 -> geo[pt*15 + n-1]
        const int tileBase = blockIdx.x * 128 + wave * 32 + t * 16;
        #pragma unroll
        for (int j = 0; j < 8; ++j) {
            const int m  = j + (hi ? 8 : 0);
            const int pt = tileBase + m;
            const float vo = c1[j];
            if (lm == 0) out[pt] = vo;
            else         out[npts + pt * 15 + (lm - 1)] = vo;
        }
        __syncthreads();
    }
}

extern "C" void kernel_launch(void* const* d_in, const int* in_sizes, int n_in,
                              void* d_out, int out_size, void* d_ws, size_t ws_size,
                              hipStream_t stream) {
    (void)n_in; (void)out_size; (void)d_ws; (void)ws_size;
    const float* x     = (const float*)d_in[0];
    const float* table = (const float*)d_in[1];
    const float* W0    = (const float*)d_in[2];
    const float* b0    = (const float*)d_in[3];
    const float* W1    = (const float*)d_in[4];
    const float* b1    = (const float*)d_in[5];
    float* out = (float*)d_out;

    const int npts   = in_sizes[0] / 3;     // 1048576
    const int blocks = npts / 128;          // 128 points per block (4 waves x 32)
    sdfnet_fused<<<blocks, 128, 0, stream>>>(x, table, W0, b0, W1, b1, out, npts);
}
